// PointNet2_7593502179647
// MI455X (gfx1250) — matrix operand order repaired and checked by reading, not compiled, not run on hardware
//
#include <hip/hip_runtime.h>
#include <float.h>

typedef _Float16 v16h __attribute__((ext_vector_type(16)));
typedef _Float16 v8h  __attribute__((ext_vector_type(8)));
typedef float    v8f  __attribute__((ext_vector_type(8)));

#define BATCH 16

// ---------------- Farthest point sampling: one block per batch ----------------
__global__ void __launch_bounds__(256) fps_kernel(const float* __restrict__ xyz, int n, int stride,
                                                  int npoint, int* __restrict__ outIdx)
{
  __shared__ float dist[8192];
  __shared__ float rv[256];
  __shared__ int   ri[256];
  int b = blockIdx.x, tid = threadIdx.x;
  const float* base = xyz + (size_t)b * n * stride;
  for (int i = tid; i < n; i += 256) dist[i] = 1e10f;
  int far = 0;
  if (tid == 0) outIdx[b * npoint] = 0;
  __syncthreads();
  for (int it = 1; it < npoint; ++it) {
    float cx = base[(size_t)far * stride + 0];
    float cy = base[(size_t)far * stride + 1];
    float cz = base[(size_t)far * stride + 2];
    float bestV = -1.0f; int bestI = 0;
    for (int i = tid; i < n; i += 256) {
      float dx = base[(size_t)i * stride + 0] - cx;
      float dy = base[(size_t)i * stride + 1] - cy;
      float dz = base[(size_t)i * stride + 2] - cz;
      float d2 = dx * dx + dy * dy + dz * dz;
      float dd = fminf(dist[i], d2);
      dist[i] = dd;
      if (dd > bestV) { bestV = dd; bestI = i; }
    }
    rv[tid] = bestV; ri[tid] = bestI;
    __syncthreads();
    for (int off = 128; off > 0; off >>= 1) {
      if (tid < off) {
        float ov = rv[tid + off]; int oi = ri[tid + off];
        if (ov > rv[tid] || (ov == rv[tid] && oi < ri[tid])) { rv[tid] = ov; ri[tid] = oi; }
      }
      __syncthreads();
    }
    far = ri[0];
    if (tid == 0) outIdx[b * npoint + it] = far;
    __syncthreads();
  }
}

// ---------------- gather: new_xyz = xyz[idx] ----------------
__global__ void gather_kernel(const float* __restrict__ xyz, int n, int stride,
                              const int* __restrict__ idx, int S, float* __restrict__ out)
{
  int i = blockIdx.x * 256 + threadIdx.x;
  if (i >= BATCH * S) return;
  int b = i / S;
  int j = idx[i];
  const float* p = xyz + ((size_t)b * n + j) * stride;
  out[(size_t)i * 3 + 0] = p[0];
  out[(size_t)i * 3 + 1] = p[1];
  out[(size_t)i * 3 + 2] = p[2];
}

// ---------------- KNN: one block per query, LDS distance array + k argmin passes --------
__global__ void __launch_bounds__(256) knn_kernel(const float* __restrict__ ref, int n, int refStride,
                                                  const float* __restrict__ query, int S, int nsample,
                                                  int* __restrict__ outIdx)
{
  __shared__ float sd[8192];
  __shared__ float rv[256];
  __shared__ int   ri[256];
  int q = blockIdx.x;              // b*S + s
  int b = q / S;
  int tid = threadIdx.x;
  const float* rbase = ref + (size_t)b * n * refStride;
  float qx = query[(size_t)q * 3 + 0];
  float qy = query[(size_t)q * 3 + 1];
  float qz = query[(size_t)q * 3 + 2];
  for (int i = tid; i < n; i += 256) {
    float dx = rbase[(size_t)i * refStride + 0] - qx;
    float dy = rbase[(size_t)i * refStride + 1] - qy;
    float dz = rbase[(size_t)i * refStride + 2] - qz;
    sd[i] = dx * dx + dy * dy + dz * dz;
  }
  __syncthreads();
  for (int j = 0; j < nsample; ++j) {
    float bestV = FLT_MAX; int bestI = n;
    for (int i = tid; i < n; i += 256) {
      float d = sd[i];
      if (d < bestV || (d == bestV && i < bestI)) { bestV = d; bestI = i; }
    }
    rv[tid] = bestV; ri[tid] = bestI;
    __syncthreads();
    for (int off = 128; off > 0; off >>= 1) {
      if (tid < off) {
        float ov = rv[tid + off]; int oi = ri[tid + off];
        if (ov < rv[tid] || (ov == rv[tid] && oi < ri[tid])) { rv[tid] = ov; ri[tid] = oi; }
      }
      __syncthreads();
    }
    if (tid == 0) { outIdx[(size_t)q * nsample + j] = ri[0]; sd[ri[0]] = FLT_MAX; }
    __syncthreads();
  }
}

// -------- build grouped f16 A matrix [rows, Kpad]; nidx==null => group_all (j=k) --------
__global__ void group_kernel(const float* __restrict__ xyz, int n, int xyzStride,
                             const float* __restrict__ newxyz,
                             const float* __restrict__ points, int cpts, int ptsStride,
                             const int* __restrict__ nidx,
                             int S, int Ksamp, int Kpad, int rows,
                             _Float16* __restrict__ A)
{
  int row = blockIdx.x * 256 + threadIdx.x;
  if (row >= rows) return;
  int b = row / (S * Ksamp);
  int rem = row - b * (S * Ksamp);
  int s = rem / Ksamp;
  int k = rem - s * Ksamp;
  int j = nidx ? nidx[row] : k;
  const float* pr = xyz + ((size_t)b * n + j) * xyzStride;
  const float* nc = newxyz + ((size_t)b * S + s) * 3;
  _Float16* a = A + (size_t)row * Kpad;
  a[0] = (_Float16)(pr[0] - nc[0]);
  a[1] = (_Float16)(pr[1] - nc[1]);
  a[2] = (_Float16)(pr[2] - nc[2]);
  int c = 3;
  if (points) {
    const float* pp = points + ((size_t)b * n + j) * ptsStride;
    for (int t = 0; t < cpts; ++t) a[3 + t] = (_Float16)pp[t];
    c = 3 + cpts;
  }
  for (; c < Kpad; ++c) a[c] = (_Float16)0.f;
}

// ---------------- mean of xyz over S points (group_all centroid) ----------------
__global__ void mean_kernel(const float* __restrict__ xyz, int S, float* __restrict__ out)
{
  int b = blockIdx.x, c = threadIdx.x;
  if (c >= 3) return;
  float s = 0.f;
  for (int i = 0; i < S; ++i) s += xyz[((size_t)b * S + i) * 3 + c];
  out[b * 3 + c] = s / (float)S;
}

// ---------------- transpose + pad weights to f16 Wt[cout, Kpad] ----------------
__global__ void wt_kernel(const float* __restrict__ w, int cin, int cout, int Kpad,
                          _Float16* __restrict__ wt)
{
  int i = blockIdx.x * 256 + threadIdx.x;
  if (i >= cout * Kpad) return;
  int co = i / Kpad, ci = i - co * Kpad;
  wt[i] = (ci < cin) ? (_Float16)w[(size_t)ci * cout + co] : (_Float16)0.f;
}

__global__ void zero_f32(float* __restrict__ p, int n)
{
  int i = blockIdx.x * 256 + threadIdx.x;
  if (i < n) p[i] = 0.f;
}

// ---------------- WMMA GEMM: Y[M,N] = A[M,K](f16) x Wt[N,K]^T(f16), f32 acc ----------------
// Each wave computes one 16-row M tile x four 16-col N tiles: the A fragment is loaded
// once per K-step and reused by 4 WMMAs (4x reduction of A global traffic; B panel is
// small and L2/WGP$-resident). Per-channel sum/sumsq fused for BatchNorm stats.
// Requires Ncol % 64 == 0 and M % 16 == 0 (true for every layer here).
__global__ void __launch_bounds__(256) gemm_wmma_kernel(const _Float16* __restrict__ A,
                                                        const _Float16* __restrict__ Wt,
                                                        float* __restrict__ Y,
                                                        float* __restrict__ sum,
                                                        float* __restrict__ ssq,
                                                        int M, int K, int Ncol)
{
  int lane = threadIdx.x & 31;
  int wave = threadIdx.x >> 5;
  int tilesN4 = Ncol >> 6;                    // groups of 4 N-tiles
  int tilesM  = M >> 4;
  int tile = blockIdx.x * 8 + wave;           // wave-uniform
  if (tile >= tilesM * tilesN4) return;       // whole wave exits: EXEC all-ones for WMMA
  int tm = tile / tilesN4, tn4 = tile - tm * tilesN4;

  // A fragment addressing (16-bit A 16x32, contiguous K to match B):
  // lanes 0-15: row M=lane, K = 0..15; lanes 16-31: K = 16..31.
  int rowA = tm * 16 + (lane & 15);
  int kA   = (lane >> 4) * 16;
  // B fragment (16-bit B 32x16): col N = lane&15; lanes 0-15 K=0..15, lanes 16-31 K=16..31.
  int colB = tn4 * 64 + (lane & 15);
  int kB   = (lane >> 4) * 16;

  const _Float16* aptr = A  + (size_t)rowA * K + kA;
  const _Float16* bptr = Wt + (size_t)colB * K + kB;
  const size_t bstride = (size_t)16 * K;      // halfs between adjacent N tiles

  v8f acc0 = {}, acc1 = {}, acc2 = {}, acc3 = {};
  for (int k0 = 0; k0 < K; k0 += 32) {
    __builtin_prefetch((const void*)(aptr + k0 + 128), 0, 0);   // global_prefetch_b8
    v16h af = *(const v16h*)(aptr + k0);        // K = k0+kA .. +15     (32B aligned)
    v16h b0 = *(const v16h*)(bptr + k0);                    // 32B aligned
    v16h b1 = *(const v16h*)(bptr + bstride + k0);
    v16h b2 = *(const v16h*)(bptr + 2 * bstride + k0);
    v16h b3 = *(const v16h*)(bptr + 3 * bstride + k0);
    acc0 = __builtin_amdgcn_wmma_f32_16x16x32_f16(false, af, false, b0, (short)0, acc0, false, false);
    acc1 = __builtin_amdgcn_wmma_f32_16x16x32_f16(false, af, false, b1, (short)0, acc1, false, false);
    acc2 = __builtin_amdgcn_wmma_f32_16x16x32_f16(false, af, false, b2, (short)0, acc2, false, false);
    acc3 = __builtin_amdgcn_wmma_f32_16x16x32_f16(false, af, false, b3, (short)0, acc3, false, false);
  }

  // C/D layout: col = lane&15, rows = 8*(lane>=16) + i
  int colbase = tn4 * 64 + (lane & 15);
  int rowbase = tm * 16 + (lane >> 4) * 8;
  v8f accs[4] = {acc0, acc1, acc2, acc3};
#pragma unroll
  for (int t = 0; t < 4; ++t) {
    int col = colbase + t * 16;
    float s = 0.f, s2 = 0.f;
#pragma unroll
    for (int i = 0; i < 8; ++i) {
      float v = accs[t][i];
      Y[(size_t)(rowbase + i) * Ncol + col] = v;
      s += v; s2 += v * v;
    }
    atomicAdd(&sum[col], s);
    atomicAdd(&ssq[col], s2);
  }
}

// ------- BN (training-mode, biased var) + ReLU; emit f16 next-layer A or f32 for pooling ----
__global__ void bn_relu_kernel(const float* __restrict__ Y,
                               const float* __restrict__ sum, const float* __restrict__ ssq,
                               const float* __restrict__ g, const float* __restrict__ be,
                               float invM, int total, int C,
                               _Float16* __restrict__ Aout, float* __restrict__ Fout)
{
  int idx = blockIdx.x * 256 + threadIdx.x;
  if (idx >= total) return;
  int c = idx % C;
  float mean = sum[c] * invM;
  float var  = fmaxf(ssq[c] * invM - mean * mean, 0.f);
  float a = g[c] * rsqrtf(var + 1e-5f);
  float v = (Y[idx] - mean) * a + be[c];
  v = fmaxf(v, 0.f);
  if (Aout) Aout[idx] = (_Float16)v;
  else      Fout[idx] = v;
}

// ---------------- max over nsample axis ----------------
__global__ void maxpool_kernel(const float* __restrict__ Yf, int BS, int Ksamp, int C,
                               float* __restrict__ out)
{
  int idx = blockIdx.x * 256 + threadIdx.x;
  if (idx >= BS * C) return;
  int bs = idx / C, c = idx - bs * C;
  const float* p = Yf + (size_t)bs * Ksamp * C + c;
  float m = -FLT_MAX;
  for (int k = 0; k < Ksamp; ++k) m = fmaxf(m, p[(size_t)k * C]);
  out[idx] = m;
}

extern "C" void kernel_launch(void* const* d_in, const int* in_sizes, int n_in,
                              void* d_out, int out_size, void* d_ws, size_t ws_size,
                              hipStream_t stream)
{
  (void)in_sizes; (void)n_in; (void)out_size; (void)ws_size;
  const float* x = (const float*)d_in[0];
  auto P = [&](int st, int l, int which) -> const float* {
    return (const float*)d_in[1 + (st * 3 + l) * 4 + which];   // w,b,gamma,beta per layer
  };

  char* wsp = (char*)d_ws;
  auto alloc = [&](size_t bytes) -> void* {
    void* p = (void*)wsp;
    wsp += (bytes + 255) & ~(size_t)255;
    return p;
  };

  int*      idx1  = (int*)     alloc((size_t)16 * 512 * 4);
  float*    nx1   = (float*)   alloc((size_t)16 * 512 * 3 * 4);
  int*      nidx1 = (int*)     alloc((size_t)16 * 512 * 32 * 4);
  int*      idx2  = (int*)     alloc((size_t)16 * 128 * 4);
  float*    nx2   = (float*)   alloc((size_t)16 * 128 * 3 * 4);
  int*      nidx2 = (int*)     alloc((size_t)16 * 128 * 64 * 4);
  float*    pts1  = (float*)   alloc((size_t)16 * 512 * 128 * 4);
  float*    pts2  = (float*)   alloc((size_t)16 * 128 * 256 * 4);
  float*    mean3 = (float*)   alloc((size_t)16 * 3 * 4);
  float*    ssum  = (float*)   alloc((size_t)1024 * 4);
  float*    ssq   = (float*)   alloc((size_t)1024 * 4);
  _Float16* wtb   = (_Float16*)alloc((size_t)1024 * 512 * 2);
  _Float16* A0    = (_Float16*)alloc((size_t)131072 * 160 * 2);   // 40 MiB (max A-ping)
  _Float16* A1    = (_Float16*)alloc((size_t)262144 * 64 * 2);    // 32 MiB (max A-pong)
  float*    Ybuf  = (float*)   alloc((size_t)262144 * 128 * 4);   // 128 MiB

  auto run_mlp = [&](int st, int Mrows, int Kpad0, int realCin,
                     const int* couts, int BS, int Ksamp, float* poolOut) {
    _Float16* cur = A0;
    int curK = Kpad0;
    int cinTrue = realCin;
    for (int l = 0; l < 3; ++l) {
      int cout = couts[l];
      int wn = cout * curK;
      wt_kernel<<<(wn + 255) / 256, 256, 0, stream>>>(P(st, l, 0), cinTrue, cout, curK, wtb);
      zero_f32<<<(cout + 255) / 256, 256, 0, stream>>>(ssum, cout);
      zero_f32<<<(cout + 255) / 256, 256, 0, stream>>>(ssq, cout);
      int tiles = (Mrows / 16) * (cout / 64);   // 1x4-tile wave blocks
      gemm_wmma_kernel<<<(tiles + 7) / 8, 256, 0, stream>>>(cur, wtb, Ybuf, ssum, ssq,
                                                            Mrows, curK, cout);
      int total = Mrows * cout;
      _Float16* nxt = (cur == A0) ? A1 : A0;
      if (l < 2) {
        bn_relu_kernel<<<(total + 255) / 256, 256, 0, stream>>>(
            Ybuf, ssum, ssq, P(st, l, 2), P(st, l, 3),
            1.0f / (float)Mrows, total, cout, nxt, nullptr);
        cur = nxt; curK = cout; cinTrue = cout;
      } else {
        bn_relu_kernel<<<(total + 255) / 256, 256, 0, stream>>>(
            Ybuf, ssum, ssq, P(st, l, 2), P(st, l, 3),
            1.0f / (float)Mrows, total, cout, nullptr, Ybuf);
        maxpool_kernel<<<(BS * cout + 255) / 256, 256, 0, stream>>>(Ybuf, BS, Ksamp, cout, poolOut);
      }
    }
  };

  // ================= Stage 1: npoint=512, nsample=32, mlp 6->64->64->128 =================
  fps_kernel<<<16, 256, 0, stream>>>(x, 8192, 6, 512, idx1);
  gather_kernel<<<(16 * 512 + 255) / 256, 256, 0, stream>>>(x, 8192, 6, idx1, 512, nx1);
  knn_kernel<<<16 * 512, 256, 0, stream>>>(x, 8192, 6, nx1, 512, 32, nidx1);
  {
    int rows = 16 * 512 * 32;
    group_kernel<<<(rows + 255) / 256, 256, 0, stream>>>(x, 8192, 6, nx1, x + 3, 3, 6, nidx1,
                                                         512, 32, 32, rows, A0);
  }
  {
    int c1[3] = {64, 64, 128};
    run_mlp(0, 16 * 512 * 32, 32, 6, c1, 16 * 512, 32, pts1);
  }

  // ================= Stage 2: npoint=128, nsample=64, mlp 131->128->128->256 =============
  fps_kernel<<<16, 256, 0, stream>>>(nx1, 512, 3, 128, idx2);
  gather_kernel<<<(16 * 128 + 255) / 256, 256, 0, stream>>>(nx1, 512, 3, idx2, 128, nx2);
  knn_kernel<<<16 * 128, 256, 0, stream>>>(nx1, 512, 3, nx2, 128, 64, nidx2);
  {
    int rows = 16 * 128 * 64;
    group_kernel<<<(rows + 255) / 256, 256, 0, stream>>>(nx1, 512, 3, nx2, pts1, 128, 128, nidx2,
                                                         128, 64, 160, rows, A0);
  }
  {
    int c2[3] = {128, 128, 256};
    run_mlp(1, 16 * 128 * 64, 160, 131, c2, 16 * 128, 64, pts2);
  }

  // ================= Stage 3: group_all, mlp 259->256->512->1024 =========================
  mean_kernel<<<16, 32, 0, stream>>>(nx2, 128, mean3);
  {
    int rows = 16 * 128;
    group_kernel<<<(rows + 255) / 256, 256, 0, stream>>>(nx2, 128, 3, mean3, pts2, 256, 256,
                                                         nullptr, 1, 128, 288, rows, A0);
  }
  {
    int c3[3] = {256, 512, 1024};
    run_mlp(2, 16 * 128, 288, 259, c3, 16, 128, (float*)d_out);
  }
}
